// PaCoLoss_69621419868363
// MI455X (gfx1250) — matrix operand order repaired and checked by reading, not compiled, not run on hardware
//
#include <hip/hip_runtime.h>
#include <hip/hip_bf16.h>
#include <math.h>

// ---------------- problem constants (match reference) ----------------
constexpr int B = 32, K = 8, D = 256;
constexpr int NNEG = 512, NCLS = 200, H = 56, W = 56;
constexpr float ALPHA = 0.2f, BETA = 0.05f, GAMMA = 0.1f;
constexpr float LAMBDA_PAC = 1.0f, ETA_SOC = 0.1f;

typedef __attribute__((ext_vector_type(16))) _Float16 v16h;
typedef __attribute__((ext_vector_type(8)))  float    v8f;

union FragU { v16h h; uint4 q[2]; };

// ---------------- workspace layout (bytes) ----------------
constexpr size_t OFF_INV   = 0;                       // 32*256*256 f32 = 8 MiB
constexpr size_t OFF_INVH  = OFF_INV   + 8388608;     // f16 copy        = 4 MiB
constexpr size_t OFF_PH    = OFF_INVH  + 4194304;     // P f16           = 256 KiB
constexpr size_t OFF_ZW    = OFF_PH    + 262144;      // zW (B*K*D) f32  = 256 KiB
constexpr size_t OFF_ZWZ   = OFF_ZW    + 262144;      // (B*K) f32
constexpr size_t OFF_DPOS  = OFF_ZWZ   + 1024;        // (B*K) f32
constexpr size_t OFF_PWP   = OFF_DPOS  + 1024;        // (B*NNEG) f32    = 64 KiB
constexpr size_t OFF_HINGE = OFF_PWP   + 65536;       // (B*K) f32
constexpr size_t OFF_SOC   = OFF_HINGE + 1024;        // 64*2 f32
constexpr size_t OFF_CE    = OFF_SOC   + 512;         // B f32
constexpr size_t OFF_WSAL  = OFF_CE    + 128;         // B f32

// ============ 1) Gauss-Jordan in-place inverse of Sigma_plus ============
__global__ void k_inv(const float* __restrict__ Sp, float* __restrict__ A,
                      _Float16* __restrict__ Ah) {
  const int b = blockIdx.x, t = threadIdx.x;
  float* M = A + (size_t)b * D * D;
  const float* S = Sp + (size_t)b * D * D;
  for (int i = t; i < D * D; i += 256) M[i] = S[i];
  __threadfence_block();
  __syncthreads();
  __shared__ float prow[D];
  for (int k = 0; k < D; ++k) {
    prow[t] = M[k * D + t];
    __syncthreads();
    const float ip = 1.0f / prow[k];
    const float f  = M[t * D + k];
    if (t == k) {
      for (int j = 0; j < D; ++j) M[k * D + j] = (j == k) ? ip : prow[j] * ip;
    } else {
      for (int j = 0; j < D; ++j) {
        float v = M[t * D + j];
        M[t * D + j] = (j == k) ? (-f * ip) : fmaf(-f * ip, prow[j], v);
      }
    }
    __threadfence_block();
    __syncthreads();
  }
  for (int i = t; i < D * D; i += 256) Ah[(size_t)b * D * D + i] = (_Float16)M[i];
}

// ============ 2) per-(b,k): zW = z^T W, zWz, d_pos; also P -> f16 ============
__global__ void k_rows(const float* __restrict__ Z1, const float* __restrict__ Z2,
                       const int* __restrict__ matching, const float* __restrict__ A,
                       const float* __restrict__ P, float* __restrict__ zW,
                       float* __restrict__ zWz, float* __restrict__ dpos,
                       _Float16* __restrict__ Ph) {
  const int bk = blockIdx.x, b = bk >> 3, kk = bk & 7, t = threadIdx.x;
  __shared__ float sz[D], sdp[D], red[256];
  const float* z = Z1 + ((size_t)b * K + kk) * D;
  const int m = matching[b * K + kk];
  const float* q = Z2 + ((size_t)b * K + m) * D;
  sz[t]  = z[t];
  sdp[t] = z[t] - q[t];
  __syncthreads();
  const float* Ab = A + (size_t)b * D * D;
  float t1 = 0.f, t2 = 0.f;
  for (int d = 0; d < D; ++d) {
    const float col = Ab[d * D + t];
    t1 = fmaf(sdp[d], col, t1);
    t2 = fmaf(sz[d],  col, t2);
  }
  zW[(size_t)bk * D + t] = t2;
  const float r1 = t1 * sdp[t], r2 = t2 * sz[t];
  red[t] = r1; __syncthreads();
  for (int s = 128; s > 0; s >>= 1) { if (t < s) red[t] += red[t + s]; __syncthreads(); }
  const float s1 = red[0]; __syncthreads();
  red[t] = r2; __syncthreads();
  for (int s = 128; s > 0; s >>= 1) { if (t < s) red[t] += red[t + s]; __syncthreads(); }
  if (t == 0) { dpos[bk] = sqrtf(fmaxf(s1, 0.f)); zWz[bk] = red[0]; }
  const int gid = bk * 256 + t;
  Ph[gid]         = (_Float16)P[gid];
  Ph[gid + 65536] = (_Float16)P[gid + 65536];
}

// ============ 3) WMMA: Y = P @ invS per batch, fused pWp[n] = <Y[n,:],P[n,:]> ====
// grid = (NNEG/64, B); block = 128 (4 wave32). Each wave owns a 16-row tile of the
// 64-negative panel. A panel staged in LDS via async global->LDS B128; W (symmetric)
// read as contiguous rows for the B fragment; pWp reduced with shfl_xor butterflies.
__global__ void __launch_bounds__(128)
k_gemm(const _Float16* __restrict__ Ah, const _Float16* __restrict__ Ph,
       float* __restrict__ pWp) {
  const int nblk = blockIdx.x;            // 0..7  (64 negatives each)
  const int b    = blockIdx.y;            // 0..31
  const int t    = threadIdx.x;           // 0..127
  const int wave = t >> 5, lane = t & 31;
  const int nbase = nblk * 64;
  __shared__ _Float16 sA[64 * 256];       // 32 KiB A panel (f16)
  // ---- async stage of the 64x256-half A panel: 2048 x b128 chunks ----
  {
    const char* gsrc = (const char*)Ph + (size_t)nbase * D * sizeof(_Float16);
    unsigned ldsbase = (unsigned)(uintptr_t)(&sA[0]);
#pragma unroll
    for (int i = 0; i < 16; ++i) {
      const int chunk = i * 128 + t;      // 0..2047
      unsigned ldsaddr = ldsbase + chunk * 16;
      const void* g = gsrc + (size_t)chunk * 16;
      asm volatile("global_load_async_to_lds_b128 %0, %1, off"
                   :: "v"(ldsaddr), "v"(g) : "memory");
    }
    asm volatile("s_wait_asynccnt 0x0" ::: "memory");
  }
  __syncthreads();

  const _Float16* Wm = Ah + (size_t)b * D * D;   // invS f16 (symmetric)
  const int l15 = lane & 15;
  const int hiA = (lane < 16) ? 0 : 8;    // A-fragment K base (ISA layout)
  const int hiB = (lane < 16) ? 0 : 8;    // B-fragment K base (ISA layout)
  const _Float16* ArowL = &sA[(wave * 16 + l15) * D];   // this lane's A row (LDS)
  float accRow[8];
#pragma unroll
  for (int r = 0; r < 8; ++r) accRow[r] = 0.f;

  for (int et = 0; et < 16; ++et) {
    const int e0 = et * 16;
    const _Float16* Wrow = Wm + (size_t)(e0 + l15) * D;  // symmetric: row e, contiguous K
    __builtin_prefetch(Wm + (size_t)(e0 + 16 + l15) * D, 0, 0);
    v8f c = {};
#pragma unroll
    for (int ks = 0; ks < 8; ++ks) {
      const int kb = ks * 32;
      FragU a, bf;
      // A 16x32 f16 fragment: halves 0..7 <-> K kb+hiA+0..7, 8..15 <-> K kb+hiA+16..23
      a.q[0] = *(const uint4*)(ArowL + kb + hiA);
      a.q[1] = *(const uint4*)(ArowL + kb + hiA + 16);
      // B 32x16 f16 fragment: halves 0..7 <-> K kb+hiB+0..7, 8..15 <-> K kb+hiB+16..23 (via W symmetry)
      bf.q[0] = *(const uint4*)(Wrow + kb + hiB);
      bf.q[1] = *(const uint4*)(Wrow + kb + hiB + 16);
      c = __builtin_amdgcn_wmma_f32_16x16x32_f16(
          false, a.h, false, bf.h, (short)0, c, false, false);
    }
    // fused pWp partial: VGPR r holds (M=r+hiA, N=e0+l15); butterfly over the
    // 16-lane half (xor 1,2,4,8) sums over e within the tile.
#pragma unroll
    for (int r = 0; r < 8; ++r) {
      const int mrow = wave * 16 + r + hiA;
      float v = c[r] * (float)sA[mrow * D + e0 + l15];
      v += __shfl_xor(v, 1);
      v += __shfl_xor(v, 2);
      v += __shfl_xor(v, 4);
      v += __shfl_xor(v, 8);
      accRow[r] += v;
    }
  }
  if (l15 == 0) {                          // lanes 0 and 16 hold the two row halves
#pragma unroll
    for (int r = 0; r < 8; ++r)
      pWp[(size_t)b * NNEG + nbase + wave * 16 + r + hiA] = accRow[r];
  }
}

// ============ 4) semi-hard mining + hinge per (b,k) ============
__global__ void k_pac(const float* __restrict__ zW, const float* __restrict__ zWz,
                      const float* __restrict__ dpos, const float* __restrict__ pWp,
                      const float* __restrict__ P, float* __restrict__ hinge) {
  const int bk = blockIdx.x, b = bk >> 3, t = threadIdx.x;
  __shared__ float srow[D], rAll[256], rSemi[256];
  srow[t] = zW[(size_t)bk * D + t];
  __syncthreads();
  const float dp = dpos[bk], zz = zWz[bk];
  float mAll = INFINITY, mSemi = INFINITY;
  for (int n = t; n < NNEG; n += 256) {
    const float* Prow = P + (size_t)n * D;
    float zp = 0.f;
    for (int e = 0; e < D; ++e) zp = fmaf(srow[e], Prow[e], zp);
    const float d2 = zz - 2.f * zp + pWp[(size_t)b * NNEG + n];
    const float dn = sqrtf(fmaxf(d2, 0.f));
    mAll = fminf(mAll, dn);
    if (dn > dp) mSemi = fminf(mSemi, dn);
  }
  rAll[t] = mAll; rSemi[t] = mSemi; __syncthreads();
  for (int s = 128; s > 0; s >>= 1) {
    if (t < s) { rAll[t] = fminf(rAll[t], rAll[t + s]); rSemi[t] = fminf(rSemi[t], rSemi[t + s]); }
    __syncthreads();
  }
  if (t == 0) {
    const float dn = (rSemi[0] < INFINITY) ? rSemi[0] : rAll[0];
    hinge[bk] = fmaxf(dp - dn + ALPHA, 0.f);
  }
}

// ============ 5) SOC: Frobenius terms ============
__global__ void k_soc(const float* __restrict__ S1, const float* __restrict__ S2,
                      const float* __restrict__ Spr, float* __restrict__ part) {
  const int t = threadIdx.x, blk = blockIdx.x;
  __shared__ float r1[256], r2[256];
  const int total = B * D * D;
  float a1 = 0.f, a2 = 0.f;
  for (int i = blk * 256 + t; i < total; i += 64 * 256) {
    const float x1 = S1[i], x2 = S2[i], xp = Spr[i];
    const float d = x1 - x2;               a1 = fmaf(d, d, a1);
    const float e = 0.5f * (x1 + x2) - xp; a2 = fmaf(e, e, a2);
  }
  r1[t] = a1; r2[t] = a2; __syncthreads();
  for (int s = 128; s > 0; s >>= 1) {
    if (t < s) { r1[t] += r1[t + s]; r2[t] += r2[t + s]; }
    __syncthreads();
  }
  if (t == 0) { part[blk * 2] = r1[0]; part[blk * 2 + 1] = r2[0]; }
}

// ============ 6) CE (log-softmax) + saliency window weights ============
__global__ void k_ce(const float* __restrict__ logits, const int* __restrict__ targets,
                     const float* __restrict__ sal, const int* __restrict__ peaks,
                     float* __restrict__ ce, float* __restrict__ wout) {
  const int b = blockIdx.x, t = threadIdx.x;
  __shared__ float red[256];
  __shared__ float wm[K];
  const float* row = logits + (size_t)b * NCLS;
  float m = -INFINITY;
  for (int j = t; j < NCLS; j += 256) m = fmaxf(m, row[j]);
  red[t] = m; __syncthreads();
  for (int s = 128; s > 0; s >>= 1) { if (t < s) red[t] = fmaxf(red[t], red[t + s]); __syncthreads(); }
  m = red[0]; __syncthreads();
  float sum = 0.f;
  for (int j = t; j < NCLS; j += 256) sum += __expf(row[j] - m);
  red[t] = sum; __syncthreads();
  for (int s = 128; s > 0; s >>= 1) { if (t < s) red[t] += red[t + s]; __syncthreads(); }
  if (t == 0) ce[b] = row[targets[b]] - m - __logf(red[0]);
  if (t < K) {
    const int ph = peaks[(b * K + t) * 2 + 0];
    const int pw = peaks[(b * K + t) * 2 + 1];
    float s = 0.f, cnt = 0.f;
    for (int dh = -2; dh <= 2; ++dh)
      for (int dw = -2; dw <= 2; ++dw) {
        const int hh = ph + dh, ww = pw + dw;
        if (hh >= 0 && hh < H && ww >= 0 && ww < W) {
          s += sal[((size_t)b * H + hh) * W + ww]; cnt += 1.f;
        }
      }
    wm[t] = s / cnt;
  }
  __syncthreads();
  if (t == 0) {
    float s = 0.f;
    for (int k2 = 0; k2 < K; ++k2) s += wm[k2];
    wout[b] = s / K;
  }
}

// ============ 7) finalize ============
__global__ void k_final(const float* __restrict__ ce, const float* __restrict__ wout,
                        const float* __restrict__ hinge, const float* __restrict__ soc,
                        float* __restrict__ out) {
  const int t = threadIdx.x;
  __shared__ float red[256];
  red[t] = hinge[t]; __syncthreads();
  for (int s = 128; s > 0; s >>= 1) { if (t < s) red[t] += red[t + s]; __syncthreads(); }
  const float Lpac = red[0] / (float)(B * K);
  if (t == 0) {
    float sce = 0.f, sw = 0.f;
    for (int b2 = 0; b2 < B; ++b2) { sce += ce[b2]; sw += wout[b2]; }
    const float base = -sce / (float)B;
    const float Lce  = base * ((1.f - GAMMA) + GAMMA * (sw / (float)B));
    float s1 = 0.f, s2 = 0.f;
    for (int i = 0; i < 64; ++i) { s1 += soc[i * 2]; s2 += soc[i * 2 + 1]; }
    const float Lsoc = s1 / (float)B + BETA * (s2 / (float)B);
    out[0] = Lce + LAMBDA_PAC * Lpac + ETA_SOC * Lsoc;
    out[1] = Lce; out[2] = Lpac; out[3] = Lsoc;
  }
}

extern "C" void kernel_launch(void* const* d_in, const int* in_sizes, int n_in,
                              void* d_out, int out_size, void* d_ws, size_t ws_size,
                              hipStream_t stream) {
  const float* logits   = (const float*)d_in[0];
  const int*   targets  = (const int*)  d_in[1];
  const float* Z1       = (const float*)d_in[2];
  const float* Z2       = (const float*)d_in[3];
  const int*   matching = (const int*)  d_in[4];
  const float* P        = (const float*)d_in[5];
  const float* S1       = (const float*)d_in[6];
  const float* S2       = (const float*)d_in[7];
  const float* Sp       = (const float*)d_in[8];
  const float* Spr      = (const float*)d_in[9];
  const float* sal      = (const float*)d_in[10];
  const int*   peaks    = (const int*)  d_in[11];

  char* ws = (char*)d_ws;
  float*     invS  = (float*)    (ws + OFF_INV);
  _Float16*  invH  = (_Float16*) (ws + OFF_INVH);
  _Float16*  Ph    = (_Float16*) (ws + OFF_PH);
  float*     zW    = (float*)    (ws + OFF_ZW);
  float*     zWz   = (float*)    (ws + OFF_ZWZ);
  float*     dpos  = (float*)    (ws + OFF_DPOS);
  float*     pWp   = (float*)    (ws + OFF_PWP);
  float*     hinge = (float*)    (ws + OFF_HINGE);
  float*     socp  = (float*)    (ws + OFF_SOC);
  float*     cew   = (float*)    (ws + OFF_CE);
  float*     wsal  = (float*)    (ws + OFF_WSAL);

  k_inv <<<B,        256, 0, stream>>>(Sp, invS, invH);
  k_rows<<<B * K,    256, 0, stream>>>(Z1, Z2, matching, invS, P, zW, zWz, dpos, Ph);
  dim3 g3(NNEG / 64, B);
  k_gemm<<<g3,       128, 0, stream>>>(invH, Ph, pWp);
  k_pac <<<B * K,    256, 0, stream>>>(zW, zWz, dpos, pWp, P, hinge);
  k_soc <<<64,       256, 0, stream>>>(S1, S2, Spr, socp);
  k_ce  <<<B,        256, 0, stream>>>(logits, targets, sal, peaks, cew, wsal);
  k_final<<<1,       256, 0, stream>>>(cew, wsal, hinge, socp, (float*)d_out);
}
